// RaZeRLinear_69269232550408
// MI455X (gfx1250) — matrix operand order repaired and checked by reading, not compiled, not run on hardware
//
#include <hip/hip_runtime.h>

typedef _Float16 h2   __attribute__((ext_vector_type(2)));
typedef _Float16 v16h __attribute__((ext_vector_type(16)));
typedef float    v8f  __attribute__((ext_vector_type(8)));
typedef unsigned u32;
typedef u32      u32x4 __attribute__((ext_vector_type(4)));

#define K_DIM   4096
#define N_DIM   14336
#define GROUP   128
#define KSPLIT  2
#define KHALF   (K_DIM / KSPLIT)            // 2048 per K-half wave
#define NGRP    (KHALF / GROUP)             // 16 scale groups per wave
#define TPB     4                           // N-tiles per block
#define LDS_A_BYTES (16 * K_DIM * 2)        // 131072: sigma-permuted A tile
#define LDS_P_OFF   LDS_A_BYTES
#define LDS_TOTAL   (LDS_A_BYTES + KSPLIT * TPB * 256 * 4)   // +8192 partials

// Dequant 8 nibbles of one int32 into 4 f16 pairs in sigma order:
// pair p = {v(p), v(p+4)} where v = signed nibble. Uses the fold
// ((w>>4p) & 0x000F000F) ^ 0x64086408  ->  halves (1032 + v) exactly
// (one v_bitop3 per pair), then fma with (s, -1032*s).
__device__ __forceinline__ void deq8(u32 w, h2 ss, h2 off, h2* bp) {
    u32 t0 = ( w        & 0x000F000Fu) ^ 0x64086408u;
    u32 t1 = ((w >> 4)  & 0x000F000Fu) ^ 0x64086408u;
    u32 t2 = ((w >> 8)  & 0x000F000Fu) ^ 0x64086408u;
    u32 t3 = ((w >> 12) & 0x000F000Fu) ^ 0x64086408u;
    bp[0] = __builtin_bit_cast(h2, t0) * ss + off;   // v_pk_fma_f16
    bp[1] = __builtin_bit_cast(h2, t1) * ss + off;
    bp[2] = __builtin_bit_cast(h2, t2) * ss + off;
    bp[3] = __builtin_bit_cast(h2, t3) * ss + off;
}

// Block: 256 threads = 8 waves = 4 N-tiles x 2 K-halves. Grid: 224 blocks.
__global__ void __launch_bounds__(256)
razer_w4a16_wmma_kernel(const _Float16* __restrict__ x,     // [16, 4096] fp16
                        const int*      __restrict__ qB,    // [512, 14336] int32
                        const _Float16* __restrict__ scl,   // [32, 14336] fp16
                        float*          __restrict__ out)   // [16, 14336] f32
{
    extern __shared__ char smem[];
    _Float16* aLds = (_Float16*)smem;
    float*    pLds = (float*)(smem + LDS_P_OFF);

    const int tid  = threadIdx.x;
    const int lane = tid & 31;
    const int warp = tid >> 5;

    // ---- stage sigma-permuted A tile (16x4096 f16, 128 KB) into LDS ----
    // within each 8-half block, position q holds x[8a + sigma(q)], sigma=[0,4,1,5,2,6,3,7]
    {
        const u32x4* xg = (const u32x4*)x;
        u32x4*       ag = (u32x4*)aLds;
        #pragma unroll
        for (int c = 0; c < 32; ++c) {
            const int ch = c * 256 + tid;           // 8192 16-byte chunks
            u32x4 d = xg[ch];                       // halves {x0..x7}
            u32x4 r;
            r[0] = __builtin_amdgcn_perm(d[2], d[0], 0x05040100u); // {x0,x4}
            r[1] = __builtin_amdgcn_perm(d[2], d[0], 0x07060302u); // {x1,x5}
            r[2] = __builtin_amdgcn_perm(d[3], d[1], 0x05040100u); // {x2,x6}
            r[3] = __builtin_amdgcn_perm(d[3], d[1], 0x07060302u); // {x3,x7}
            ag[ch] = r;                             // ds_store_b128
        }
    }
    __syncthreads();

    const int half_sel = lane >> 4;                 // A, B: K sub-range; acc: row half
    const int lane16   = lane & 15;
    const int kh       = warp >> 2;                 // K-half 0/1
    const int tb       = warp & 3;                  // N-tile within block
    const int n0       = (blockIdx.x * TPB + tb) * 16;

    const int b_col = n0 + lane16;
    const int b_kof = half_sel * 16;
    const _Float16* aBase = aLds + lane16 * K_DIM + kh * KHALF + half_sel * 16;

    v8f acc = {};

    for (int g = 0; g < NGRP; ++g) {                // 16 groups of K=128
        const int gAbs = kh * NGRP + g;
        _Float16 s = __builtin_nontemporal_load(&scl[gAbs * N_DIM + b_col]);
        h2 ss;  ss[0] = s;  ss[1] = s;
        h2 off; off[0] = (_Float16)(-1032.0f) * s; off[1] = off[0];

        const int* wp = qB + ((gAbs * GROUP + b_kof) >> 3) * N_DIM + b_col;

        #pragma unroll
        for (int kk = 0; kk < GROUP; kk += 32) {
            // A fragment from LDS (sigma-permuted): two ds_load_b128
            v16h a;
            *((u32x4*)&a)       = *(const u32x4*)(aBase + g * GROUP + kk);
            *(((u32x4*)&a) + 1) = *(const u32x4*)(aBase + g * GROUP + kk + 8);

            // 2 dwords of packed weights (NT: streamed once)
            u32 w0 = (u32)__builtin_nontemporal_load(wp + ((kk >> 3)    ) * N_DIM);
            u32 w1 = (u32)__builtin_nontemporal_load(wp + ((kk >> 3) + 1) * N_DIM);

            v16h b;
            deq8(w0, ss, off, (h2*)&b);
            deq8(w1, ss, off, ((h2*)&b) + 4);

            acc = __builtin_amdgcn_wmma_f32_16x16x32_f16(
                false, a, false, b, (short)0, acc, false, false);
        }

        if (g + 1 < NGRP) {                         // next group's weight lines
            __builtin_prefetch(wp + 16 * N_DIM, 0, 1);   // global_prefetch_b8
        }
    }

    // ---- combine the two K-half partials via LDS (deterministic order) ----
    float* pw = pLds + (kh * TPB + tb) * 256;
    #pragma unroll
    for (int i = 0; i < 8; ++i) {
        pw[(half_sel * 8 + i) * 16 + lane16] = acc[i];
    }
    __syncthreads();

    {
        const int row = tid >> 4, col = tid & 15;
        #pragma unroll
        for (int t = 0; t < TPB; ++t) {
            float v = pLds[t * 256 + tid] + pLds[(TPB + t) * 256 + tid];
            out[row * N_DIM + (blockIdx.x * TPB + t) * 16 + col] = v;
        }
    }
}

extern "C" void kernel_launch(void* const* d_in, const int* in_sizes, int n_in,
                              void* d_out, int out_size, void* d_ws, size_t ws_size,
                              hipStream_t stream) {
    const _Float16* x   = (const _Float16*)d_in[0];   // (1,16,4096) fp16
    const int*      qB  = (const int*)d_in[1];        // (512,14336) int32
    const _Float16* scl = (const _Float16*)d_in[2];   // (32,14336) fp16
    float*          out = (float*)d_out;              // (1,16,14336)

    dim3 grid(N_DIM / 16 / TPB);     // 224 blocks
    dim3 block(256);                 // 8 waves: 4 N-tiles x 2 K-halves
    razer_w4a16_wmma_kernel<<<grid, block, LDS_TOTAL, stream>>>(x, qB, scl, out);
}
